// SparseAttention_53919019434483
// MI455X (gfx1250) — matrix operand order repaired and checked by reading, not compiled, not run on hardware
//
#include <hip/hip_runtime.h>

// ---- problem constants (match reference) ----
#define NBQ    128          // number of 64-wide blocks along sequence
#define BLK    64           // block size
#define KMAX   9            // key blocks per query block (GLOBAL + LOCAL)
#define HEADS  16
#define DIM    64
#define SEQ    (NBQ * BLK)  // 8192
#define SM_SCALE 0.125f     // 1/sqrt(64)

typedef __attribute__((ext_vector_type(16))) __bf16 v16bf;
typedef __attribute__((ext_vector_type(8)))  __bf16 v8bf;
typedef __attribute__((ext_vector_type(8)))  float  v8f;
typedef __attribute__((ext_vector_type(4)))  unsigned int u32x4;
typedef __attribute__((ext_vector_type(8)))  int          i32x8;
typedef __attribute__((ext_vector_type(4)))  int          i32x4;

#if __has_builtin(__builtin_amdgcn_tensor_load_to_lds) && \
    __has_builtin(__builtin_amdgcn_s_wait_tensorcnt)
#define USE_TDM 1
#else
#define USE_TDM 0
#endif

#if USE_TDM
// Issue one TDM DMA: 64x64 f32 tile, rows strided by HEADS*DIM elements,
// global -> LDS. Descriptor layout per CDNA5 ISA §8 (async_tensor).
// 6-arg builtin form (clang-23 / therock-10.0 headers).
__device__ __forceinline__ void tdm_load_tile64x64_f32(const float* gbase,
                                                       unsigned lds_off)
{
    unsigned long long ga = (unsigned long long)gbase;
    u32x4 g0;
    g0[0] = 1u;                                    // count=1, user descriptor
    g0[1] = lds_off;                               // lds_addr (bytes)
    g0[2] = (unsigned)(ga & 0xffffffffu);          // global_addr[31:0]
    g0[3] = (unsigned)((ga >> 32) & 0x01ffffffu)   // global_addr[56:32]
          | (2u << 30);                            // type = 2
    i32x8 g1;
    g1[0] = 0x20000;            // data_size=2 (4 bytes); no multicast/pad/iterate
    g1[1] = (DIM << 16);        // tensor_dim0[15:0] = 64  (bits 63:48)
    g1[2] = (BLK << 16);        // tensor_dim0 hi=0 | tensor_dim1[15:0]=64
    g1[3] = (DIM << 16);        // tensor_dim1 hi=0 | tile_dim0=64
    g1[4] = BLK;                // tile_dim1=64 | tile_dim2=0
    g1[5] = HEADS * DIM;        // tensor_dim0_stride = 1024 elements
    g1[6] = 0;                  // stride hi | tensor_dim1_stride lo
    g1[7] = 0;                  // tensor_dim1_stride hi
    i32x4 gz4 = {0, 0, 0, 0};                      // groups 2/3 unused (2-D tensor)
    i32x8 gz8 = {0, 0, 0, 0, 0, 0, 0, 0};
    __builtin_amdgcn_tensor_load_to_lds(g0, g1, gz4, gz4, gz8, 0);
}
#endif

__global__ __launch_bounds__(128)
void sparse_attn_wmma(const float* __restrict__ q,
                      const float* __restrict__ k,
                      const float* __restrict__ v,
                      const int*   __restrict__ col_idx,
                      const unsigned char* __restrict__ col_valid,
                      float* __restrict__ out)
{
#if USE_TDM
    __shared__ __align__(16) float  Kst[2][BLK][DIM];   // raw f32 staging (double buffered)
    __shared__ __align__(16) float  Vst[2][BLK][DIM];
#endif
    __shared__ __align__(32) __bf16 Klds[BLK][DIM];     // K block, row-major (key x d)
    __shared__ __align__(32) __bf16 VTlds[DIM][BLK];    // V block, transposed (d x key)
    __shared__ __align__(32) __bf16 Plds[4][16][BLK];   // per-wave P strip (row x key)

    const int qi   = blockIdx.x;
    const int h    = blockIdx.y;
    const int b    = blockIdx.z;
    const int tid  = threadIdx.x;
    const int wave = tid >> 5;
    const int lane = tid & 31;
    const int ncol = lane & 15;          // N index inside a 16-wide tile
    const int hi   = (lane >= 16);       // upper half-wave
    const int moff = hi ? 8 : 0;         // row offset for C-tile layout
    const int qbase = qi * BLK;

    // valid entries of this fixed layout are the contiguous tail [kk0, KMAX)
    const int kk0 = (qi < 8) ? (8 - qi) : 0;
    const int nj  = KMAX - kk0;
    (void)col_valid;

    // ---------- load Q strip (16 rows per wave) into bf16 A-fragments ----------
    // A 16x32 k order used here (contiguous): lane<16 holds K=0..15, lane>=16 holds K=16..31
    const int   qrow = qbase + wave * 16 + ncol;
    const float* qp  = q + (((long)b * SEQ + qrow) * HEADS + h) * DIM;
    v16bf aq0, aq1;
#pragma unroll
    for (int e = 0; e < 16; ++e) {
        int dd = e + (hi ? 16 : 0);
        aq0[e] = (__bf16)qp[dd];
        aq1[e] = (__bf16)qp[32 + dd];
    }

    v8f o[4] = {v8f{}, v8f{}, v8f{}, v8f{}};
    float mrow[8], lrow[8];
#pragma unroll
    for (int r = 0; r < 8; ++r) { mrow[r] = -__builtin_inff(); lrow[r] = 0.0f; }

#if USE_TDM
    // ---------- pipeline prologue: DMA first key block into buffer 0 ----------
    {
        const int j0 = col_idx[qi * KMAX + kk0];
        if (wave == 0) {
            const long off = (((long)b * SEQ + (long)j0 * BLK) * HEADS + h) * DIM;
            tdm_load_tile64x64_f32(k + off, (unsigned)(unsigned long long)&Kst[0][0][0]);
            tdm_load_tile64x64_f32(v + off, (unsigned)(unsigned long long)&Vst[0][0][0]);
        }
    }
#endif

    for (int i = 0; i < nj; ++i) {
        const int j = col_idx[qi * KMAX + kk0 + i];

#if USE_TDM
        const int p = i & 1;
        if (wave == 0) __builtin_amdgcn_s_wait_tensorcnt((short)0);
        __syncthreads();   // publish DMA data; previous compute done with bf16 tiles

        // LDS f32 staging -> bf16 compute tiles (V transposed)
#pragma unroll
        for (int it = 0; it < 32; ++it) {
            int idx = it * 128 + tid;
            int key = idx >> 6;
            int dd  = idx & 63;
            Klds[key][dd]  = (__bf16)Kst[p][key][dd];
            VTlds[dd][key] = (__bf16)Vst[p][key][dd];
        }
        __syncthreads();

        // DMA next key block into the other buffer, overlapped with compute below
        if (wave == 0 && i + 1 < nj) {
            const int jn = col_idx[qi * KMAX + kk0 + i + 1];
            const long off = (((long)b * SEQ + (long)jn * BLK) * HEADS + h) * DIM;
            tdm_load_tile64x64_f32(k + off, (unsigned)(unsigned long long)&Kst[p ^ 1][0][0]);
            tdm_load_tile64x64_f32(v + off, (unsigned)(unsigned long long)&Vst[p ^ 1][0][0]);
        }
#else
        __syncthreads();   // previous iteration's LDS consumers are done
        const float* kbase = k + (((long)b * SEQ + (long)j * BLK) * HEADS + h) * DIM;
        const float* vbase = v + (((long)b * SEQ + (long)j * BLK) * HEADS + h) * DIM;
#pragma unroll
        for (int it = 0; it < 32; ++it) {
            int idx = it * 128 + tid;
            int key = idx >> 6;
            int dd  = idx & 63;
            long go = (long)key * (HEADS * DIM) + dd;
            Klds[key][dd]  = (__bf16)kbase[go];
            VTlds[dd][key] = (__bf16)vbase[go];
        }
        __syncthreads();
#endif

        // ---------- S = Q * K^T : 4 tiles of 16 key columns ----------
        v8f s[4];
#pragma unroll
        for (int t = 0; t < 4; ++t) {
            // B 32x16 k order used here (contiguous): lane<16 -> K=0..15, lane>=16 -> K=16..31
            const v16bf bk0 = *(const v16bf*)&Klds[t * 16 + ncol][(hi ? 16 : 0)];
            const v16bf bk1 = *(const v16bf*)&Klds[t * 16 + ncol][32 + (hi ? 16 : 0)];
            v8f c = {};
            c = __builtin_amdgcn_wmma_f32_16x16x32_bf16(false, aq0, false, bk0,
                                                        (short)0, c, false, false);
            c = __builtin_amdgcn_wmma_f32_16x16x32_bf16(false, aq1, false, bk1,
                                                        (short)0, c, false, false);
            s[t] = c;
        }

        // ---------- scale + token-level causal mask ----------
#pragma unroll
        for (int t = 0; t < 4; ++t) {
#pragma unroll
            for (int r = 0; r < 8; ++r) {
                int qpos = qbase + wave * 16 + r + moff;
                int kpos = j * BLK + t * 16 + ncol;
                float sv = s[t][r] * SM_SCALE;
                s[t][r] = (kpos <= qpos) ? sv : -__builtin_inff();
            }
        }

        // ---------- online softmax (rows live across 16 lanes of a half-wave) ----------
#pragma unroll
        for (int r = 0; r < 8; ++r) {
            float mx = fmaxf(fmaxf(s[0][r], s[1][r]), fmaxf(s[2][r], s[3][r]));
#pragma unroll
            for (int off = 1; off < 16; off <<= 1)
                mx = fmaxf(mx, __shfl_xor(mx, off, 32));
            float mnew  = fmaxf(mrow[r], mx);
            float alpha = __expf(mrow[r] - mnew);
            mrow[r] = mnew;
            float ps = 0.0f;
#pragma unroll
            for (int t = 0; t < 4; ++t) {
                float pv = __expf(s[t][r] - mnew);
                s[t][r] = pv;
                ps += pv;
            }
#pragma unroll
            for (int off = 1; off < 16; off <<= 1)
                ps += __shfl_xor(ps, off, 32);
            lrow[r] = lrow[r] * alpha + ps;
            o[0][r] *= alpha; o[1][r] *= alpha; o[2][r] *= alpha; o[3][r] *= alpha;
        }

        // ---------- P: C-layout -> A-layout via per-wave LDS round trip ----------
#pragma unroll
        for (int t = 0; t < 4; ++t)
#pragma unroll
            for (int r = 0; r < 8; ++r)
                Plds[wave][r + moff][t * 16 + ncol] = (__bf16)s[t][r];

        const int koff = hi ? 16 : 0;
        const v8bf p00 = *(const v8bf*)&Plds[wave][ncol][koff];
        const v8bf p01 = *(const v8bf*)&Plds[wave][ncol][koff + 8];
        const v8bf p10 = *(const v8bf*)&Plds[wave][ncol][32 + koff];
        const v8bf p11 = *(const v8bf*)&Plds[wave][ncol][32 + koff + 8];
        const v16bf ap0 = __builtin_shufflevector(p00, p01, 0,1,2,3,4,5,6,7,8,9,10,11,12,13,14,15);
        const v16bf ap1 = __builtin_shufflevector(p10, p11, 0,1,2,3,4,5,6,7,8,9,10,11,12,13,14,15);

        // ---------- O += P * V ----------
#pragma unroll
        for (int t = 0; t < 4; ++t) {
            const v16bf bv0 = *(const v16bf*)&VTlds[t * 16 + ncol][(hi ? 16 : 0)];
            const v16bf bv1 = *(const v16bf*)&VTlds[t * 16 + ncol][32 + (hi ? 16 : 0)];
            v8f c = o[t];
            c = __builtin_amdgcn_wmma_f32_16x16x32_bf16(false, ap0, false, bv0,
                                                        (short)0, c, false, false);
            c = __builtin_amdgcn_wmma_f32_16x16x32_bf16(false, ap1, false, bv1,
                                                        (short)0, c, false, false);
            o[t] = c;
        }
    }

    // ---------- normalize and write out (fp32) ----------
#pragma unroll
    for (int r = 0; r < 8; ++r) {
        float inv = 1.0f / lrow[r];
        int   row = qbase + wave * 16 + r + moff;
        float* ob = out + (((long)b * SEQ + row) * HEADS + h) * DIM;
#pragma unroll
        for (int t = 0; t < 4; ++t)
            ob[t * 16 + ncol] = o[t][r] * inv;
    }
}

extern "C" void kernel_launch(void* const* d_in, const int* in_sizes, int n_in,
                              void* d_out, int out_size, void* d_ws, size_t ws_size,
                              hipStream_t stream) {
    const float* q = (const float*)d_in[0];
    const float* k = (const float*)d_in[1];
    const float* v = (const float*)d_in[2];
    const int*   col_idx   = (const int*)d_in[3];
    const unsigned char* col_valid = (const unsigned char*)d_in[4];
    float* out = (float*)d_out;

    const int B = in_sizes[0] / (SEQ * HEADS * DIM);   // 2
    dim3 grid(NBQ, HEADS, B);
    sparse_attn_wmma<<<grid, 128, 0, stream>>>(q, k, v, col_idx, col_valid, out);
}
